// Net_45947560132797
// MI455X (gfx1250) — matrix operand order repaired and checked by reading, not compiled, not run on hardware
//
#include <hip/hip_runtime.h>
#include <math.h>

typedef __attribute__((ext_vector_type(16))) __bf16 v16bf;
typedef __attribute__((ext_vector_type(8)))  float  v8f;

union FragBf { v16bf v; uint4 q[2]; };
union FragF  { v8f   v; float f[8]; };

#define TM 64
#define TN 64
#define TK 32

__device__ __forceinline__ unsigned lds_off(const void* p) {
    return (unsigned)(unsigned long long)p;   // generic LDS addr: low 32 bits = LDS offset
}

// 4 async b128 lane-copies: 16B x2 for one A row-half, 16B x2 for one B row-half.
__device__ __forceinline__ void stage_async(unsigned ldsA, const __bf16* gA,
                                            unsigned ldsB, const __bf16* gB)
{
    asm volatile(
        "global_load_async_to_lds_b128 %0, %2, off\n\t"
        "global_load_async_to_lds_b128 %0, %2, off offset:16\n\t"
        "global_load_async_to_lds_b128 %1, %3, off\n\t"
        "global_load_async_to_lds_b128 %1, %3, off offset:16"
        :: "v"(ldsA), "v"(ldsB), "v"(gA), "v"(gB) : "memory");
}

// 8 async b128 lane-copies: two full 64B rows (offset applies to LDS+global).
__device__ __forceinline__ void stage8(unsigned l0, const __bf16* s0,
                                       unsigned l1, const __bf16* s1)
{
    asm volatile(
        "global_load_async_to_lds_b128 %0, %2, off\n\t"
        "global_load_async_to_lds_b128 %0, %2, off offset:16\n\t"
        "global_load_async_to_lds_b128 %0, %2, off offset:32\n\t"
        "global_load_async_to_lds_b128 %0, %2, off offset:48\n\t"
        "global_load_async_to_lds_b128 %1, %3, off\n\t"
        "global_load_async_to_lds_b128 %1, %3, off offset:16\n\t"
        "global_load_async_to_lds_b128 %1, %3, off offset:32\n\t"
        "global_load_async_to_lds_b128 %1, %3, off offset:48"
        :: "v"(l0), "v"(l1), "v"(s0), "v"(s1) : "memory");
}

// ---------------------------------------------------------------------------
// Generic bf16 WMMA GEMM (prenet / postnet): C[M,N] = A[M,K] @ Bt[N,K]^T (+bias),
// optional leaky-relu, stores f32 and/or bf16. B pre-transposed, N padded.
// ---------------------------------------------------------------------------
__global__ __launch_bounds__(128) void gemm_bf16(
    const __bf16* __restrict__ A, const __bf16* __restrict__ Bt,
    const float* __restrict__ bias,
    float* __restrict__ Cf, __bf16* __restrict__ Cb,
    int M, int N, int K, long lda, long ldb, long ldc, int leaky)
{
    __shared__ __align__(16) __bf16 As[2][TM][TK];
    __shared__ __align__(16) __bf16 Bs[2][TN][TK];

    const int tid  = threadIdx.x;
    const int lane = tid & 31;
    const int wave = tid >> 5;
    const int wm   = (wave & 1) * 32;
    const int wn   = (wave >> 1) * 32;
    const int hlf  = lane >> 4;
    const int l15  = lane & 15;

    const int rowBase = blockIdx.y * TM;
    const int colBase = blockIdx.x * TN;

    const int srow = tid >> 1;
    const int soff = (tid & 1) * 16;
    const __bf16* gA = A  + (long)(rowBase + srow) * lda + soff;
    const __bf16* gB = Bt + (long)(colBase + srow) * ldb + soff;

    FragF acc[2][2] = {};
    const int ntiles = K / TK;

    stage_async(lds_off(&As[0][srow][soff]), gA, lds_off(&Bs[0][srow][soff]), gB);

    for (int it = 0; it < ntiles; ++it) {
        const int cur = it & 1;
        if (it + 1 < ntiles) {
            const long ko = (long)(it + 1) * TK;
            stage_async(lds_off(&As[cur ^ 1][srow][soff]), gA + ko,
                        lds_off(&Bs[cur ^ 1][srow][soff]), gB + ko);
            asm volatile("s_wait_asynccnt 4" ::: "memory");
        } else {
            asm volatile("s_wait_asynccnt 0" ::: "memory");
        }
        __syncthreads();

        FragBf a[2], b[2];
        for (int i = 0; i < 2; ++i) {
            const __bf16* ap = &As[cur][wm + i * 16 + l15][0];
            a[i].q[0] = *(const uint4*)(ap + hlf * 16);
            a[i].q[1] = *(const uint4*)(ap + hlf * 16 + 8);
        }
        for (int j = 0; j < 2; ++j) {
            const __bf16* bp = &Bs[cur][wn + j * 16 + l15][hlf * 16];
            b[j].q[0] = *(const uint4*)(bp);
            b[j].q[1] = *(const uint4*)(bp + 8);
        }
        for (int i = 0; i < 2; ++i)
            for (int j = 0; j < 2; ++j)
                acc[i][j].v = __builtin_amdgcn_wmma_f32_16x16x32_bf16(
                    false, a[i].v, false, b[j].v, (short)0, acc[i][j].v, false, false);
        __syncthreads();
    }

    for (int i = 0; i < 2; ++i) {
        const int r0 = rowBase + wm + i * 16 + hlf * 8;
        for (int j = 0; j < 2; ++j) {
            const int col = colBase + wn + j * 16 + l15;
            if (col >= N) continue;
            const float bsum = bias ? bias[col] : 0.0f;
            for (int r = 0; r < 8; ++r) {
                float v = acc[i][j].f[r] + bsum;
                if (leaky) v = (v > 0.0f) ? v : 0.1f * v;
                const long o = (long)(r0 + r) * ldc + col;
                if (Cf) Cf[o] = v;
                if (Cb) Cb[o] = (__bf16)v;
            }
        }
    }
}

// ---------------------------------------------------------------------------
// Fused GRU step (one layer): both gate GEMMs + cell update in one kernel.
// Grid: (H/64, Bb/64) = (16,2) blocks x 256 threads (8 waves).
// Block owns batch rows [r0,r0+64) x hidden cols [c0,c0+64); accumulates the
// six 64x64 tiles {x,h} x {r,z,n}. Wave w: path s=w&1, quadrant (wm,wn);
// 12 WMMAs per wave per K-step. Hidden state double-buffered by caller.
// Dynamic LDS: [0,64K) staging (2 bufs x 8 tiles x 4KB), reused as 96KB f32
// scratch for the cross-wave gate exchange in the epilogue.
// ---------------------------------------------------------------------------
__global__ __launch_bounds__(256) void gru_step_fused(
    const __bf16* __restrict__ Ax, long ldax,        // layer input [128,1024]
    const __bf16* __restrict__ Ah,                   // h_in bf16 [128,1024]
    const __bf16* __restrict__ BxT,                  // WxT [3072][1024]
    const __bf16* __restrict__ BhT,                  // WhT [3072][1024]
    const float* __restrict__ bx, const float* __restrict__ bh,   // [3072]
    const float* __restrict__ hfin,                  // h_in f32 [128,1024]
    float* __restrict__ hfout, __bf16* __restrict__ hbout,        // h_out
    __bf16* __restrict__ outv, long outStride)       // nullable
{
    extern __shared__ __align__(16) unsigned char smem[];
    __bf16 (*As)[2][64][TK]    = (__bf16 (*)[2][64][TK])smem;              // [buf][s][row][k]
    __bf16 (*Bs)[2][3][64][TK] = (__bf16 (*)[2][3][64][TK])(smem + 16384); // [buf][s][g][row][k]
    float* scr = (float*)smem;                                             // [s][g][64][64]

    const int tid  = threadIdx.x;
    const int lane = tid & 31;
    const int wave = tid >> 5;            // 0..7
    const int s    = wave & 1;            // 0: x-path, 1: h-path
    const int wm   = ((wave >> 1) & 1) * 32;
    const int wn   = ((wave >> 2) & 1) * 32;
    const int hlf  = lane >> 4;
    const int l15  = lane & 15;

    const int c0 = blockIdx.x * 64;       // hidden-column slice
    const int r0 = blockIdx.y * 64;       // batch-row tile

    // ---- staging assignment: tile tau (0..7), two 64B rows per thread ----
    const int tau  = tid >> 5;
    const int trow = (lane) * 2;          // rows trow, trow+1 of the tile
    const __bf16* gb0; const __bf16* gb1;
    unsigned lb00, lb01, lb10, lb11;      // [buf][rr]
    {
        const int ss = (tau - 2) / 3, gg = (tau - 2) % 3;
        if (tau == 0) {
            gb0 = Ax + (long)(r0 + trow) * ldax;
            gb1 = Ax + (long)(r0 + trow + 1) * ldax;
        } else if (tau == 1) {
            gb0 = Ah + (long)(r0 + trow) * 1024;
            gb1 = Ah + (long)(r0 + trow + 1) * 1024;
        } else {
            const __bf16* B = ss ? BhT : BxT;
            gb0 = B + ((long)(gg << 10) + c0 + trow) * 1024;
            gb1 = B + ((long)(gg << 10) + c0 + trow + 1) * 1024;
        }
        if (tau < 2) {
            lb00 = lds_off(&As[0][tau][trow][0]);     lb01 = lds_off(&As[0][tau][trow + 1][0]);
            lb10 = lds_off(&As[1][tau][trow][0]);     lb11 = lds_off(&As[1][tau][trow + 1][0]);
        } else {
            lb00 = lds_off(&Bs[0][ss][gg][trow][0]);  lb01 = lds_off(&Bs[0][ss][gg][trow + 1][0]);
            lb10 = lds_off(&Bs[1][ss][gg][trow][0]);  lb11 = lds_off(&Bs[1][ss][gg][trow + 1][0]);
        }
    }

    FragF acc[3][2][2] = {};
    constexpr int NTILES = 1024 / TK;     // 32

    stage8(lb00, gb0, lb01, gb1);

    for (int it = 0; it < NTILES; ++it) {
        const int cur = it & 1;
        if (it + 1 < NTILES) {
            const long ko = (long)(it + 1) * TK;
            if (cur) stage8(lb00, gb0 + ko, lb01, gb1 + ko);
            else     stage8(lb10, gb0 + ko, lb11, gb1 + ko);
            asm volatile("s_wait_asynccnt 8" ::: "memory");
        } else {
            asm volatile("s_wait_asynccnt 0" ::: "memory");
        }
        __syncthreads();

        FragBf a[2];
        for (int i = 0; i < 2; ++i) {
            const __bf16* ap = &As[cur][s][wm + i * 16 + l15][0];
            a[i].q[0] = *(const uint4*)(ap + hlf * 16);
            a[i].q[1] = *(const uint4*)(ap + hlf * 16 + 8);
        }
        for (int g = 0; g < 3; ++g) {
            FragBf b[2];
            for (int j = 0; j < 2; ++j) {
                const __bf16* bp = &Bs[cur][s][g][wn + j * 16 + l15][hlf * 16];
                b[j].q[0] = *(const uint4*)(bp);
                b[j].q[1] = *(const uint4*)(bp + 8);
            }
            for (int i = 0; i < 2; ++i)
                for (int j = 0; j < 2; ++j)
                    acc[g][i][j].v = __builtin_amdgcn_wmma_f32_16x16x32_bf16(
                        false, a[i].v, false, b[j].v, (short)0, acc[g][i][j].v, false, false);
        }
        __syncthreads();
    }

    // ---- epilogue: exchange the six gate tiles through LDS scratch ----
    for (int g = 0; g < 3; ++g)
        for (int i = 0; i < 2; ++i)
            for (int j = 0; j < 2; ++j)
                for (int r = 0; r < 8; ++r) {
                    const int row = wm + i * 16 + hlf * 8 + r;
                    const int col = wn + j * 16 + l15;
                    scr[(((s * 3 + g) * 64) + row) * 64 + col] = acc[g][i][j].f[r];
                }
    __syncthreads();

    // ---- GRU cell update: 16 elements per thread ----
    const int urow = tid >> 2;                 // 0..63
    const int ucol0 = (tid & 3) * 16;
    const int brow = r0 + urow;
    for (int e = 0; e < 16; ++e) {
        const int col  = ucol0 + e;
        const int hcol = c0 + col;
        const float rx = scr[((0 * 64) + urow) * 64 + col] + bx[hcol];
        const float zx = scr[((1 * 64) + urow) * 64 + col] + bx[1024 + hcol];
        const float nx = scr[((2 * 64) + urow) * 64 + col] + bx[2048 + hcol];
        const float rh = scr[((3 * 64) + urow) * 64 + col] + bh[hcol];
        const float zh = scr[((4 * 64) + urow) * 64 + col] + bh[1024 + hcol];
        const float nh = scr[((5 * 64) + urow) * 64 + col] + bh[2048 + hcol];
        const float hold = hfin[(long)brow * 1024 + hcol];
        const float r  = 1.0f / (1.0f + __expf(-(rx + rh)));
        const float zz = 1.0f / (1.0f + __expf(-(zx + zh)));
        const float n  = tanhf(nx + r * nh);
        const float hn = (1.0f - zz) * n + zz * hold;
        hfout[(long)brow * 1024 + hcol] = hn;
        hbout[(long)brow * 1024 + hcol] = (__bf16)hn;
        if (outv) outv[(long)brow * outStride + hcol] = (__bf16)hn;
    }
}

// transpose + convert + pad: in f32 [K_in, N_in] -> out bf16 [Np, Kp]
__global__ void transpose_cvt(const float* __restrict__ in, __bf16* __restrict__ out,
                              int K_in, int N_in, int Kp, int Np)
{
    const long i = (long)blockIdx.x * blockDim.x + threadIdx.x;
    if (i >= (long)Np * Kp) return;
    const int n = (int)(i / Kp);
    const int k = (int)(i - (long)n * Kp);
    out[i] = (k < K_in && n < N_in) ? (__bf16)in[(long)k * N_in + n] : (__bf16)0.0f;
}

// pad last dim: [rows, kin] f32 -> [rows, kout] bf16 (zero-padded)
__global__ void pad_last(const float* __restrict__ in, __bf16* __restrict__ out,
                         long rows, int kin, int kout)
{
    const long i = (long)blockIdx.x * blockDim.x + threadIdx.x;
    if (i >= rows * (long)kout) return;
    const long r = i / kout;
    const int  k = (int)(i - r * kout);
    out[i] = (k < kin) ? (__bf16)in[r * kin + k] : (__bf16)0.0f;
}

extern "C" void kernel_launch(void* const* d_in, const int* in_sizes, int n_in,
                              void* d_out, int out_size, void* d_ws, size_t ws_size,
                              hipStream_t stream)
{
    const float* x     = (const float*)d_in[0];
    const float* Wpre  = (const float*)d_in[1];
    const float* bpre  = (const float*)d_in[2];
    const float* Wx    = (const float*)d_in[3];
    const float* Wh    = (const float*)d_in[4];
    const float* bx    = (const float*)d_in[5];
    const float* bh    = (const float*)d_in[6];
    const float* Wpost = (const float*)d_in[7];
    const float* bpost = (const float*)d_in[8];
    float* out = (float*)d_out;

    constexpr int  Bb = 128, T = 512, F = 88, H = 1024, L = 2, G3 = 3 * H;
    constexpr int  FP = 96, FNP = 128;
    constexpr long BT = (long)Bb * T;
    constexpr long BH = (long)Bb * H;

    char* p = (char*)d_ws;
    auto carve = [&](size_t bytes) { char* r = p; p += (bytes + 255) & ~(size_t)255; return r; };
    __bf16* xp     = (__bf16*)carve(BT * FP * 2);
    __bf16* wpreT  = (__bf16*)carve((size_t)H * FP * 2);
    __bf16* wxT    = (__bf16*)carve((size_t)L * G3 * H * 2);
    __bf16* whT    = (__bf16*)carve((size_t)L * G3 * H * 2);
    __bf16* wpostT = (__bf16*)carve((size_t)FNP * H * 2);
    __bf16* res    = (__bf16*)carve((size_t)BT * H * 2);
    __bf16* outs   = (__bf16*)carve((size_t)BT * H * 2);
    float*  hf     = (float*)carve((size_t)L * 2 * BH * 4);   // [l][phase][B][H]
    __bf16* hbf    = (__bf16*)carve((size_t)L * 2 * BH * 2);

    // ---- convert / pad / pre-transpose to bf16 ----
    {
        long n = BT * FP;
        pad_last<<<dim3((unsigned)((n + 255) / 256)), 256, 0, stream>>>(x, xp, BT, F, FP);
        long m = (long)H * FP;
        transpose_cvt<<<dim3((unsigned)((m + 255) / 256)), 256, 0, stream>>>(
            Wpre, wpreT, F, H, FP, H);
        long nw = (long)G3 * H;
        for (int l = 0; l < L; ++l) {
            transpose_cvt<<<dim3((unsigned)((nw + 255) / 256)), 256, 0, stream>>>(
                Wx + (long)l * H * G3, wxT + (long)l * G3 * H, H, G3, H, G3);
            transpose_cvt<<<dim3((unsigned)((nw + 255) / 256)), 256, 0, stream>>>(
                Wh + (long)l * H * G3, whT + (long)l * G3 * H, H, G3, H, G3);
        }
        long np = (long)FNP * H;
        transpose_cvt<<<dim3((unsigned)((np + 255) / 256)), 256, 0, stream>>>(
            Wpost, wpostT, H, F, H, FNP);
    }
    hipMemsetAsync(hf,  0, (size_t)L * 2 * BH * 4, stream);
    hipMemsetAsync(hbf, 0, (size_t)L * 2 * BH * 2, stream);

    // ---- prenet: res = leaky_relu(xp @ Wpre + bpre) (bf16) ----
    gemm_bf16<<<dim3(H / TN, (unsigned)(BT / TM), 1), 128, 0, stream>>>(
        xp, wpreT, bpre, nullptr, res, (int)BT, H, FP, FP, FP, H, 1);

    // ---- recurrence: one fused kernel per (step, layer) ----
    const size_t fusedLds = 98304;   // 96KB dynamic LDS
    for (int t = 0; t < T; ++t) {
        const int ph = t & 1;
        for (int l = 0; l < L; ++l) {
            float*  hin  = hf  + ((long)l * 2 + ph) * BH;
            float*  hout = hf  + ((long)l * 2 + (ph ^ 1)) * BH;
            __bf16* bin  = hbf + ((long)l * 2 + ph) * BH;
            __bf16* bout = hbf + ((long)l * 2 + (ph ^ 1)) * BH;
            const __bf16* Ax = (l == 0) ? (res + (long)t * H)
                                        : (hbf + (0L * 2 + (ph ^ 1)) * BH); // layer0 h_out
            const long ldax = (l == 0) ? (long)T * H : (long)H;
            gru_step_fused<<<dim3(H / 64, Bb / 64), 256, fusedLds, stream>>>(
                Ax, ldax, bin,
                wxT + (long)l * G3 * H, whT + (long)l * G3 * H,
                bx + (long)l * G3, bh + (long)l * G3,
                hin, hout, bout,
                (l == L - 1) ? (outs + (long)t * H) : nullptr, (long)T * H);
        }
    }

    // ---- postnet: out = outs @ Wpost + bpost (f32) ----
    gemm_bf16<<<dim3(FNP / TN, (unsigned)(BT / TM), 1), 128, 0, stream>>>(
        outs, wpostT, bpost, out, nullptr, (int)BT, F, H, H, H, F, 0);

    (void)in_sizes; (void)n_in; (void)out_size; (void)ws_size;
}
